// PointTransformerLayer_58677843198013
// MI455X (gfx1250) — matrix operand order repaired and checked by reading, not compiled, not run on hardware
//
#include <hip/hip_runtime.h>
#include <hip/hip_bf16.h>
#include <math.h>

#define DIM     384
#define HEADS   6
#define HD      64
#define MLPH    768
#define BATCH   4
#define SEQ     2048
#define ROWS    (BATCH*SEQ)          // 8192 tokens
#define QKVC    (3*DIM)              // 1152

typedef _Float16 h8_t  __attribute__((ext_vector_type(8)));
typedef _Float16 h16_t __attribute__((ext_vector_type(16)));
typedef float    f8_t  __attribute__((ext_vector_type(8)));

// ---------------------------------------------------------------------------
// Fragment helpers (CDNA5 wave32 WMMA 16x16x32 f16 layouts, CDNA5 ISA)
//  A-frag : lane L holds row M=L&15; 16B at K = 16*(L>>4) and 16*(L>>4)+8
//  B-frag : lane L holds col N=L&15 of N-major (transposed) operand;
//           16B at K = 16*(L>>4) and 16*(L>>4)+8
//  C/D    : VGPR g holds row g+8*(L>>4), col L&15
// ---------------------------------------------------------------------------
static __device__ __forceinline__ h16_t ld2(const _Float16* p0, const _Float16* p1) {
  h8_t lo = *reinterpret_cast<const h8_t*>(p0);
  h8_t hi = *reinterpret_cast<const h8_t*>(p1);
  return __builtin_shufflevector(lo, hi, 0,1,2,3,4,5,6,7,8,9,10,11,12,13,14,15);
}

static __device__ __forceinline__ f8_t wmma16(h16_t a, h16_t b, f8_t c) {
  return __builtin_amdgcn_wmma_f32_16x16x32_f16(false, a, false, b, (short)0, c,
                                                false, false);
}

// Core: one wave computes a 16(row) x 64(col) fp32 tile of A[M,K] * Bt[N,K]^T
static __device__ __forceinline__ void gemm_tile(
    const _Float16* __restrict__ A, int lda,
    const _Float16* __restrict__ Bt, int ldb,
    int rowBase, int colBase, int Kdim, f8_t acc[4])
{
  const int lane = threadIdx.x & 31;
  const int hf   = lane >> 4;
  const int l16  = lane & 15;
  const _Float16* arow = A + (size_t)(rowBase + l16) * lda;
  for (int k = 0; k < Kdim; k += 32) {
    h16_t a = ld2(arow + k + 16*hf, arow + k + 16*hf + 8);
    #pragma unroll
    for (int t = 0; t < 4; ++t) {
      const _Float16* brow = Bt + (size_t)(colBase + 16*t + l16) * ldb + k + 16*hf;
      h16_t b = ld2(brow, brow + 8);
      acc[t] = wmma16(a, b, acc[t]);
    }
  }
}

// ---------------------------------------------------------------------------
// LayerNorm (fp32 in) -> f16 out.  One wave per token row (D=384).
// ---------------------------------------------------------------------------
__global__ __launch_bounds__(256) void ln_kernel(
    const float* __restrict__ x, const float* __restrict__ g,
    const float* __restrict__ bta, _Float16* __restrict__ out)
{
  const int w    = threadIdx.x >> 5;
  const int lane = threadIdx.x & 31;
  const int row  = blockIdx.x * 8 + w;
  const float* xr = x + (size_t)row * DIM;
  float v[12], s = 0.f, s2 = 0.f;
  #pragma unroll
  for (int i = 0; i < 12; ++i) { v[i] = xr[lane + 32*i]; s += v[i]; s2 += v[i]*v[i]; }
  #pragma unroll
  for (int off = 16; off > 0; off >>= 1) {
    s  += __shfl_xor(s,  off);
    s2 += __shfl_xor(s2, off);
  }
  const float mu   = s * (1.f/DIM);
  const float var  = s2 * (1.f/DIM) - mu*mu;
  const float rstd = rsqrtf(var + 1e-5f);
  #pragma unroll
  for (int i = 0; i < 12; ++i) {
    const int c = lane + 32*i;
    out[(size_t)row*DIM + c] = (_Float16)((v[i]-mu)*rstd*g[c] + bta[c]);
  }
}

// fp32 [R,C]  ->  f16 transposed [C,R]  (weights -> N-major for B-frags)
__global__ __launch_bounds__(256) void castT_kernel(
    const float* __restrict__ in, _Float16* __restrict__ out, int R, int C)
{
  const int idx = blockIdx.x * 256 + threadIdx.x;
  if (idx >= R*C) return;
  const int r = idx / C, c = idx % C;
  out[(size_t)c*R + r] = (_Float16)in[idx];
}

// ---------------------------------------------------------------------------
// QKV GEMM: [8192,384]f16 x [1152,384]f16^T; epilogue scatters
//   q (pre-scaled by 1/sqrt(Hd)) and k as [b,h,n,hd] f16, v as V^T [b,h,hd,n] f16
// ---------------------------------------------------------------------------
__global__ __launch_bounds__(256) void qkv_gemm_kernel(
    const _Float16* __restrict__ A, const _Float16* __restrict__ Bt,
    const float* __restrict__ bias,
    _Float16* __restrict__ qO, _Float16* __restrict__ kO, _Float16* __restrict__ vO)
{
  const int w = threadIdx.x >> 5, lane = threadIdx.x & 31;
  const int hf = lane >> 4, l16 = lane & 15;
  const int rowBase = (blockIdx.x * 8 + w) * 16;
  const int colBase = blockIdx.y * 64;
  f8_t acc[4] = {};
  gemm_tile(A, DIM, Bt, DIM, rowBase, colBase, DIM, acc);
  #pragma unroll
  for (int t = 0; t < 4; ++t) {
    const int col = colBase + 16*t + l16;
    const int j   = col / DIM;        // 0=q 1=k 2=v (uniform in wave)
    const int d   = col % DIM;
    const int h   = d >> 6, hd = d & 63;
    #pragma unroll
    for (int g = 0; g < 8; ++g) {
      const int row = rowBase + g + 8*hf;
      const int b = row >> 11, n = row & (SEQ-1);
      const float val = acc[t][g] + bias[col];
      const size_t idx = (((size_t)b*HEADS + h)*SEQ + n)*HD + hd;
      if (j == 0)      qO[idx] = (_Float16)(val * 0.125f);   // 1/sqrt(64)
      else if (j == 1) kO[idx] = (_Float16)val;
      else vO[(((size_t)b*HEADS + h)*HD + hd)*SEQ + n] = (_Float16)val;
    }
  }
}

// ---------------------------------------------------------------------------
// Fused flash attention with exp(-||pi-pj||^2) bias.
// wave = 16 query rows; loop keys in chunks of 32; 8 WMMAs / chunk.
// ---------------------------------------------------------------------------
__global__ __launch_bounds__(256) void attn_kernel(
    const _Float16* __restrict__ Qg, const _Float16* __restrict__ Kg,
    const _Float16* __restrict__ Vt, const float* __restrict__ pos,
    _Float16* __restrict__ out)
{
  __shared__ __align__(16) _Float16 smem[8][16][40];   // per-wave 16x32 P tile (+pad)
  const int w = threadIdx.x >> 5, lane = threadIdx.x & 31;
  const int hf = lane >> 4, l16 = lane & 15;
  const int bh = blockIdx.y, b = bh / HEADS, h = bh % HEADS;
  const int qbase = (blockIdx.x * 8 + w) * 16;

  const _Float16* Q = Qg + (size_t)bh * SEQ * HD;
  const _Float16* K = Kg + (size_t)bh * SEQ * HD;
  const _Float16* V = Vt + (size_t)bh * HD * SEQ;     // [hd][n]

  // Q A-frags (reused over the whole key loop)
  const _Float16* qrow = Q + (size_t)(qbase + l16) * HD;
  const h16_t aq0 = ld2(qrow + 16*hf,      qrow + 16*hf + 8);   // k 0..31
  const h16_t aq1 = ld2(qrow + 32 + 16*hf, qrow + 40 + 16*hf);   // k 32..63

  float qpx[8], qpy[8], qpz[8];
  #pragma unroll
  for (int g = 0; g < 8; ++g) {
    const int r = qbase + g + 8*hf;
    const float* p = pos + ((size_t)b*SEQ + r)*3;
    qpx[g] = p[0]; qpy[g] = p[1]; qpz[g] = p[2];
  }

  float m[8], l[8];
  f8_t o0 = {}, o1 = {}, o2 = {}, o3 = {};
  #pragma unroll
  for (int g = 0; g < 8; ++g) { m[g] = -1e30f; l[g] = 0.f; }

  for (int kb = 0; kb < SEQ; kb += 32) {
    float s0[8], s1[8];
    // ---- S = Q K^T (+ distance bias), two 16-key sub-tiles ----
    {
      const int kc = kb + l16;
      const _Float16* kr = K + (size_t)kc * HD;
      f8_t s = {};
      s = wmma16(aq0, ld2(kr + 16*hf,      kr + 16*hf + 8),      s);
      s = wmma16(aq1, ld2(kr + 32 + 16*hf, kr + 32 + 16*hf + 8), s);
      const float* kp = pos + ((size_t)b*SEQ + kc)*3;
      const float kx = kp[0], ky = kp[1], kz = kp[2];
      #pragma unroll
      for (int g = 0; g < 8; ++g) {
        const float dx = qpx[g]-kx, dy = qpy[g]-ky, dz = qpz[g]-kz;
        s0[g] = s[g] + __expf(-(dx*dx + dy*dy + dz*dz));
      }
    }
    {
      const int kc = kb + 16 + l16;
      const _Float16* kr = K + (size_t)kc * HD;
      f8_t s = {};
      s = wmma16(aq0, ld2(kr + 16*hf,      kr + 16*hf + 8),      s);
      s = wmma16(aq1, ld2(kr + 32 + 16*hf, kr + 32 + 16*hf + 8), s);
      const float* kp = pos + ((size_t)b*SEQ + kc)*3;
      const float kx = kp[0], ky = kp[1], kz = kp[2];
      #pragma unroll
      for (int g = 0; g < 8; ++g) {
        const float dx = qpx[g]-kx, dy = qpy[g]-ky, dz = qpz[g]-kz;
        s1[g] = s[g] + __expf(-(dx*dx + dy*dy + dz*dz));
      }
    }
    // ---- online softmax over this 32-key chunk ----
    float corr[8];
    #pragma unroll
    for (int g = 0; g < 8; ++g) {
      float cm = fmaxf(s0[g], s1[g]);
      cm = fmaxf(cm, __shfl_xor(cm, 1));
      cm = fmaxf(cm, __shfl_xor(cm, 2));
      cm = fmaxf(cm, __shfl_xor(cm, 4));
      cm = fmaxf(cm, __shfl_xor(cm, 8));
      const float mn = fmaxf(m[g], cm);
      const float c  = __expf(m[g] - mn);
      const float p0 = __expf(s0[g] - mn);
      const float p1 = __expf(s1[g] - mn);
      float rs = p0 + p1;
      rs += __shfl_xor(rs, 1);
      rs += __shfl_xor(rs, 2);
      rs += __shfl_xor(rs, 4);
      rs += __shfl_xor(rs, 8);
      l[g] = l[g]*c + rs; m[g] = mn; corr[g] = c;
      const int r = g + 8*hf;                 // C-layout row within tile
      smem[w][r][l16]      = (_Float16)p0;
      smem[w][r][16 + l16] = (_Float16)p1;
    }
    #pragma unroll
    for (int g = 0; g < 8; ++g) {
      o0[g] *= corr[g]; o1[g] *= corr[g]; o2[g] *= corr[g]; o3[g] *= corr[g];
    }
    // wave-local LDS RAW: CDNA5 split DS counter
    asm volatile("s_wait_dscnt 0" ::: "memory");
    // P as A-frag (16q x 32k) from LDS
    const _Float16* pr = &smem[w][l16][0];
    const h16_t pa = ld2(pr + 16*hf, pr + 16*hf + 8);
    // V B-frags from V^T[hd][n]; O += P V
    const _Float16* v0 = V + (size_t)l16 * SEQ + kb + 16*hf;
    o0 = wmma16(pa, ld2(v0,            v0 + 8),            o0);
    o1 = wmma16(pa, ld2(v0 + 16*SEQ,   v0 + 16*SEQ + 8),   o1);
    o2 = wmma16(pa, ld2(v0 + 32*SEQ,   v0 + 32*SEQ + 8),   o2);
    o3 = wmma16(pa, ld2(v0 + 48*SEQ,   v0 + 48*SEQ + 8),   o3);
  }

  #pragma unroll
  for (int g = 0; g < 8; ++g) {
    const float inv = 1.0f / l[g];
    const int r = qbase + g + 8*hf;
    const size_t base = ((size_t)b*SEQ + r)*DIM + (size_t)h*HD;
    out[base +      l16] = (_Float16)(o0[g]*inv);
    out[base + 16 + l16] = (_Float16)(o1[g]*inv);
    out[base + 32 + l16] = (_Float16)(o2[g]*inv);
    out[base + 48 + l16] = (_Float16)(o3[g]*inv);
  }
}

// ---------------------------------------------------------------------------
// GEMM + bias + fp32 residual -> fp32 out   (proj and mlp2)
// ---------------------------------------------------------------------------
__global__ __launch_bounds__(256) void gemm_bias_res_kernel(
    const _Float16* __restrict__ A, const _Float16* __restrict__ Bt,
    const float* __restrict__ bias, const float* __restrict__ res,
    float* __restrict__ out, int Kdim, int Ncols)
{
  const int w = threadIdx.x >> 5, lane = threadIdx.x & 31;
  const int hf = lane >> 4, l16 = lane & 15;
  const int rowBase = (blockIdx.x * 8 + w) * 16;
  const int colBase = blockIdx.y * 64;
  f8_t acc[4] = {};
  gemm_tile(A, Kdim, Bt, Kdim, rowBase, colBase, Kdim, acc);
  #pragma unroll
  for (int t = 0; t < 4; ++t) {
    const int col = colBase + 16*t + l16;
    #pragma unroll
    for (int g = 0; g < 8; ++g) {
      const int row = rowBase + g + 8*hf;
      const size_t idx = (size_t)row*Ncols + col;
      out[idx] = acc[t][g] + bias[col] + res[idx];
    }
  }
}

// GEMM + bias + exact GELU -> f16 out  (mlp1)
__global__ __launch_bounds__(256) void gemm_bias_gelu_kernel(
    const _Float16* __restrict__ A, const _Float16* __restrict__ Bt,
    const float* __restrict__ bias, _Float16* __restrict__ out,
    int Kdim, int Ncols)
{
  const int w = threadIdx.x >> 5, lane = threadIdx.x & 31;
  const int hf = lane >> 4, l16 = lane & 15;
  const int rowBase = (blockIdx.x * 8 + w) * 16;
  const int colBase = blockIdx.y * 64;
  f8_t acc[4] = {};
  gemm_tile(A, Kdim, Bt, Kdim, rowBase, colBase, Kdim, acc);
  #pragma unroll
  for (int t = 0; t < 4; ++t) {
    const int col = colBase + 16*t + l16;
    #pragma unroll
    for (int g = 0; g < 8; ++g) {
      const int row = rowBase + g + 8*hf;
      const float v = acc[t][g] + bias[col];
      const float gl = 0.5f * v * (1.0f + erff(v * 0.70710678118f));
      out[(size_t)row*Ncols + col] = (_Float16)gl;
    }
  }
}

// ---------------------------------------------------------------------------
extern "C" void kernel_launch(void* const* d_in, const int* in_sizes, int n_in,
                              void* d_out, int out_size, void* d_ws, size_t ws_size,
                              hipStream_t stream)
{
  const float* x      = (const float*)d_in[0];
  const float* pos    = (const float*)d_in[1];
  const float* qkv_w  = (const float*)d_in[2];
  const float* qkv_b  = (const float*)d_in[3];
  const float* proj_w = (const float*)d_in[4];
  const float* proj_b = (const float*)d_in[5];
  const float* mlp_w1 = (const float*)d_in[6];
  const float* mlp_b1 = (const float*)d_in[7];
  const float* mlp_w2 = (const float*)d_in[8];
  const float* mlp_b2 = (const float*)d_in[9];
  const float* ln1_g  = (const float*)d_in[10];
  const float* ln1_b  = (const float*)d_in[11];
  const float* ln2_g  = (const float*)d_in[12];
  const float* ln2_b  = (const float*)d_in[13];
  float* outp = (float*)d_out;

  char* p = (char*)d_ws;
  auto alloc = [&](size_t bytes) -> char* {
    char* r = p; p += (bytes + 255) & ~(size_t)255; return r;
  };
  _Float16* ln1_16 = (_Float16*)alloc((size_t)ROWS*DIM*2);
  _Float16* qkvWt  = (_Float16*)alloc((size_t)QKVC*DIM*2);
  _Float16* q16    = (_Float16*)alloc((size_t)BATCH*HEADS*SEQ*HD*2);
  _Float16* k16    = (_Float16*)alloc((size_t)BATCH*HEADS*SEQ*HD*2);
  _Float16* vT16   = (_Float16*)alloc((size_t)BATCH*HEADS*SEQ*HD*2);
  _Float16* attn16 = (_Float16*)alloc((size_t)ROWS*DIM*2);
  _Float16* projWt = (_Float16*)alloc((size_t)DIM*DIM*2);
  float*    x1     = (float*)   alloc((size_t)ROWS*DIM*4);
  _Float16* ln2_16 = (_Float16*)alloc((size_t)ROWS*DIM*2);
  _Float16* w1t    = (_Float16*)alloc((size_t)MLPH*DIM*2);
  _Float16* w2t    = (_Float16*)alloc((size_t)DIM*MLPH*2);
  _Float16* h1     = (_Float16*)alloc((size_t)ROWS*MLPH*2);

  // 1) LN1 -> f16
  ln_kernel<<<dim3(ROWS/8), 256, 0, stream>>>(x, ln1_g, ln1_b, ln1_16);
  // 2) weight casts / transposes -> f16 N-major
  castT_kernel<<<dim3((DIM*QKVC)/256), 256, 0, stream>>>(qkv_w, qkvWt, DIM, QKVC);
  castT_kernel<<<dim3((DIM*DIM)/256),  256, 0, stream>>>(proj_w, projWt, DIM, DIM);
  castT_kernel<<<dim3((DIM*MLPH)/256), 256, 0, stream>>>(mlp_w1, w1t, DIM, MLPH);
  castT_kernel<<<dim3((MLPH*DIM)/256), 256, 0, stream>>>(mlp_w2, w2t, MLPH, DIM);
  // 3) QKV projection, scatter q/k/V^T
  qkv_gemm_kernel<<<dim3(ROWS/128, QKVC/64), 256, 0, stream>>>(
      ln1_16, qkvWt, qkv_b, q16, k16, vT16);
  // 4) fused attention
  attn_kernel<<<dim3(SEQ/128, BATCH*HEADS), 256, 0, stream>>>(
      q16, k16, vT16, pos, attn16);
  // 5) output projection + residual -> x1
  gemm_bias_res_kernel<<<dim3(ROWS/128, DIM/64), 256, 0, stream>>>(
      attn16, projWt, proj_b, x, x1, DIM, DIM);
  // 6) LN2 -> f16
  ln_kernel<<<dim3(ROWS/8), 256, 0, stream>>>(x1, ln2_g, ln2_b, ln2_16);
  // 7) MLP up + GELU
  gemm_bias_gelu_kernel<<<dim3(ROWS/128, MLPH/64), 256, 0, stream>>>(
      ln2_16, w1t, mlp_b1, h1, DIM, MLPH);
  // 8) MLP down + bias + residual -> out
  gemm_bias_res_kernel<<<dim3(ROWS/128, DIM/64), 256, 0, stream>>>(
      h1, w2t, mlp_b2, x1, outp, MLPH, DIM);
}
